// SparseAutoencoder_80779744903230
// MI455X (gfx1250) — matrix operand order repaired and checked by reading, not compiled, not run on hardware
//
#include <hip/hip_runtime.h>

#define BATCH   8192
#define IN_DIM  1024
#define LAT     16384
#define TOPK    32
#define GAMMA_C 1.0f
#define BETA_C  1.0f
#define STARTC  32

typedef __attribute__((ext_vector_type(16))) __bf16 v16bf;
typedef __attribute__((ext_vector_type(8)))  __bf16 v8bf;
typedef __attribute__((ext_vector_type(8)))  float  v8f;

typedef int v4i __attribute__((vector_size(16)));   // matches builtin param type

#if defined(__has_builtin)
#if __has_builtin(__builtin_amdgcn_global_load_async_to_lds_b128)
#define HAVE_ASYNC_BUILTIN 1
#endif
#if __has_builtin(__builtin_amdgcn_s_wait_asynccnt)
#define HAVE_WAIT_ASYNC_BUILTIN 1
#endif
#endif

__device__ __forceinline__ void async_copy16(const void* g, void* l) {
#ifdef HAVE_ASYNC_BUILTIN
    __builtin_amdgcn_global_load_async_to_lds_b128((v4i*)g, (v4i*)l, 0, 0);
#else
    unsigned int lofs = (unsigned int)(unsigned long long)l;
    asm volatile("global_load_async_to_lds_b128 %0, %1, off"
                 :: "v"(lofs), "v"((unsigned long long)g)
                 : "memory");
#endif
}

__device__ __forceinline__ void wait_async0() {
#ifdef HAVE_WAIT_ASYNC_BUILTIN
    __builtin_amdgcn_s_wait_asynccnt(0);
#else
    asm volatile("s_wait_asynccnt 0x0" ::: "memory");
#endif
}

__device__ __forceinline__ unsigned short f32_to_bf16(float f) {
    unsigned int u = __float_as_uint(f);
    u += 0x7FFFu + ((u >> 16) & 1u);      // round-to-nearest-even
    return (unsigned short)(u >> 16);
}

// ---------------------------------------------------------------- cvt f32->bf16
__global__ __launch_bounds__(256) void cvt_f32_bf16(const float* __restrict__ src,
                                                    unsigned short* __restrict__ dst,
                                                    int n4) {
    int i = blockIdx.x * 256 + threadIdx.x;
    if (i >= n4) return;
    float4 v = ((const float4*)src)[i];
    unsigned int lo = ((unsigned int)f32_to_bf16(v.y) << 16) | f32_to_bf16(v.x);
    unsigned int hi = ((unsigned int)f32_to_bf16(v.w) << 16) | f32_to_bf16(v.z);
    ((uint2*)dst)[i] = make_uint2(lo, hi);
}

// ---------------------------------------------------------------- W_dec transpose
// Wd: [IN_DIM][LAT] row-major  ->  WdT: [LAT][IN_DIM] row-major
__global__ void transpose_wdec(const float* __restrict__ Wd, float* __restrict__ WdT) {
    __shared__ float tile[32][33];
    int bx = blockIdx.x;            // along LAT
    int by = blockIdx.y;            // along IN_DIM
    int tx = threadIdx.x;           // 0..31
    int ty = threadIdx.y;           // 0..7
#pragma unroll
    for (int k = 0; k < 4; k++) {
        int d = by * 32 + ty + 8 * k;
        int l = bx * 32 + tx;
        tile[ty + 8 * k][tx] = Wd[(size_t)d * LAT + l];
    }
    __syncthreads();
#pragma unroll
    for (int k = 0; k < 4; k++) {
        int l = bx * 32 + ty + 8 * k;
        int d = by * 32 + tx;
        WdT[(size_t)l * IN_DIM + d] = tile[tx][ty + 8 * k];
    }
}

// ---------------------------------------------------------------- encoder GEMM (bf16 WMMA)
// Z[m,n] = jumprelu( sum_k x[m,k] * W_enc[n,k] + b_enc[n] )
// Double-buffered LDS; tiles staged with CDNA5 async global->LDS DMA (ASYNCcnt).
#define KT  32
#define LDT 40   // 32 + 8 ushort pad -> 80B row stride, 16B aligned chunks

__global__ __launch_bounds__(256) void encoder_wmma(
    const unsigned short* __restrict__ Abf,   // x    bf16 [BATCH][IN_DIM]
    const unsigned short* __restrict__ Bbf,   // Wenc bf16 [LAT][IN_DIM]
    const float* __restrict__ b_enc,
    float* __restrict__ Z)                    // [BATCH][LAT]
{
    __shared__ unsigned short As[2][128 * LDT];
    __shared__ unsigned short Bs[2][128 * LDT];

    const int n0 = blockIdx.x * 128;
    const int m0 = blockIdx.y * 128;
    const int t  = threadIdx.x;
    const int w  = t >> 5;
    const int lane = t & 31;
    const int hh = lane >> 4;     // lane half
    const int ll = lane & 15;
    const int wm = w & 3;         // 32-row slab within 128
    const int wn = w >> 2;        // 64-col slab within 128

    // issue async copies of one 128x32 bf16 tile pair (512 x 16B chunks each)
    auto stage = [&](int buf, int kk) {
#pragma unroll
        for (int it = 0; it < 2; it++) {
            int lin = t + 256 * it;
            int row = lin >> 2;
            int ch  = lin & 3;
            async_copy16(Abf + (size_t)(m0 + row) * IN_DIM + kk + ch * 8,
                         &As[buf][row * LDT + ch * 8]);
            async_copy16(Bbf + (size_t)(n0 + row) * IN_DIM + kk + ch * 8,
                         &Bs[buf][row * LDT + ch * 8]);
        }
    };

    v8f acc[2][4];
#pragma unroll
    for (int i = 0; i < 2; i++)
#pragma unroll
        for (int j = 0; j < 4; j++)
            acc[i][j] = (v8f){0.f, 0.f, 0.f, 0.f, 0.f, 0.f, 0.f, 0.f};

    stage(0, 0);
    int cur = 0;

#pragma unroll 1
    for (int kk = 0; kk < IN_DIM; kk += KT) {
        wait_async0();            // my async copies into `cur` are done
        __syncthreads();          // everyone's copies are done; everyone done reading cur^1
        if (kk + KT < IN_DIM) stage(cur ^ 1, kk + KT);   // overlap DMA with WMMA

        v16bf a[2], b[4];
#pragma unroll
        for (int i = 0; i < 2; i++) {
            int r = wm * 32 + i * 16 + ll;
            v8bf lo = *(const v8bf*)(&As[cur][r * LDT + 8 * hh]);
            v8bf hi = *(const v8bf*)(&As[cur][r * LDT + 16 + 8 * hh]);
            a[i] = __builtin_shufflevector(lo, hi, 0,1,2,3,4,5,6,7,8,9,10,11,12,13,14,15);
        }
#pragma unroll
        for (int j = 0; j < 4; j++) {
            int n = wn * 64 + j * 16 + ll;
            v8bf lo = *(const v8bf*)(&Bs[cur][n * LDT + 8 * hh]);
            v8bf hi = *(const v8bf*)(&Bs[cur][n * LDT + 16 + 8 * hh]);
            b[j] = __builtin_shufflevector(lo, hi, 0,1,2,3,4,5,6,7,8,9,10,11,12,13,14,15);
        }
#pragma unroll
        for (int i = 0; i < 2; i++)
#pragma unroll
            for (int j = 0; j < 4; j++)
                acc[i][j] = __builtin_amdgcn_wmma_f32_16x16x32_bf16(
                    false, a[i], false, b[j], (short)0, acc[i][j], false, false);
        cur ^= 1;
    }

    // epilogue: bias + JumpReLU
#pragma unroll
    for (int i = 0; i < 2; i++) {
#pragma unroll
        for (int j = 0; j < 4; j++) {
            int col = n0 + wn * 64 + j * 16 + ll;
            float be = b_enc[col];
#pragma unroll
            for (int r = 0; r < 8; r++) {
                int rowm = m0 + wm * 32 + i * 16 + 8 * hh + r;
                float v = acc[i][j][r] + be;
                float o = (v > 0.f ? v : 0.f) + (v > GAMMA_C ? BETA_C : 0.f);
                Z[(size_t)rowm * LAT + col] = o;
            }
        }
    }
}

// ---------------------------------------------------------------- top-k mask (per row)
__global__ __launch_bounds__(256) void topk_mask(float* __restrict__ Z,
                                                 float* __restrict__ selv_g,
                                                 int* __restrict__ seli_g) {
    __shared__ short keys[LAT];        // 32KB: monotonic bf16 keys (z >= 0)
    __shared__ int wval[8], wpack[8];
    __shared__ int seli[TOPK];
    __shared__ int cur_pack;

    const int row = blockIdx.x;
    const int t = threadIdx.x;
    const int w = t >> 5;
    const int lane = t & 31;
    float* zr = Z + (size_t)row * LAT;

    // fill keys for own strided set; -2 = excluded (cols 0..31 and last)
#pragma unroll 1
    for (int j = 0; j < 64; j++) {
        int i = t + 256 * j;
        short k;
        if (i < STARTC || i == LAT - 1) k = -2;
        else k = (short)(__float_as_uint(zr[i]) >> 16);   // z>=0: order-preserving
        keys[i] = k;
    }
    // local argmax over own set (no sync needed: own elements only)
    int kb = -3, ib = 0;
#pragma unroll 1
    for (int j = 0; j < 64; j++) {
        int i = t + 256 * j;
        int k = keys[i];
        if (k > kb) { kb = k; ib = i; }
    }
    int pk = (t << 16) | ib;
    __syncthreads();

#pragma unroll 1
    for (int s = 0; s < TOPK; s++) {
        int v = kb, p = pk;
        // wave32 XOR-swizzle argmax reduction (group-of-32 ds_swizzle)
        { int ov = __builtin_amdgcn_ds_swizzle(v, 0x401F), op = __builtin_amdgcn_ds_swizzle(p, 0x401F);
          if (ov > v || (ov == v && op < p)) { v = ov; p = op; } }
        { int ov = __builtin_amdgcn_ds_swizzle(v, 0x201F), op = __builtin_amdgcn_ds_swizzle(p, 0x201F);
          if (ov > v || (ov == v && op < p)) { v = ov; p = op; } }
        { int ov = __builtin_amdgcn_ds_swizzle(v, 0x101F), op = __builtin_amdgcn_ds_swizzle(p, 0x101F);
          if (ov > v || (ov == v && op < p)) { v = ov; p = op; } }
        { int ov = __builtin_amdgcn_ds_swizzle(v, 0x081F), op = __builtin_amdgcn_ds_swizzle(p, 0x081F);
          if (ov > v || (ov == v && op < p)) { v = ov; p = op; } }
        { int ov = __builtin_amdgcn_ds_swizzle(v, 0x041F), op = __builtin_amdgcn_ds_swizzle(p, 0x041F);
          if (ov > v || (ov == v && op < p)) { v = ov; p = op; } }
        if (lane == 0) { wval[w] = v; wpack[w] = p; }
        __syncthreads();
        if (t == 0) {
            int bv = wval[0], bp = wpack[0];
            for (int q = 1; q < 8; q++)
                if (wval[q] > bv || (wval[q] == bv && wpack[q] < bp)) { bv = wval[q]; bp = wpack[q]; }
            int idx = bp & 0xFFFF;
            seli[s] = idx;
            keys[idx] = -1;           // -1 = selected
            cur_pack = bp;
        }
        __syncthreads();
        if ((cur_pack >> 16) == t) {  // only winner rescans its 64 elements
            kb = -3; ib = 0;
            for (int j = 0; j < 64; j++) {
                int i = t + 256 * j;
                int k = keys[i];
                if (k > kb) { kb = k; ib = i; }
            }
            pk = (t << 16) | ib;
        }
        __syncthreads();
    }

    // exact selected values -> workspace lists for decoder
    if (t < TOPK) {
        int idx = seli[t];
        selv_g[(size_t)row * TOPK + t] = zr[idx];
        seli_g[(size_t)row * TOPK + t] = idx;
    }
    // zero every non-selected entry; selected (-1) keep their value (no race)
#pragma unroll 1
    for (int j = 0; j < 64; j++) {
        int i = t + 256 * j;
        if (keys[i] != -1) zr[i] = 0.f;
    }
}

// ---------------------------------------------------------------- sparse decoder
__global__ __launch_bounds__(256) void decoder_gather(
    const float* __restrict__ selv_g, const int* __restrict__ seli_g,
    const float* __restrict__ WdT,    // [LAT][IN_DIM]
    const float* __restrict__ b_dec,
    float* __restrict__ Xrec)         // [BATCH][IN_DIM]
{
    const int row = blockIdx.x;
    const int t = threadIdx.x;          // 256 threads x float4 = 1024 dims
    float4 acc = ((const float4*)b_dec)[t];
    const float* vv = selv_g + (size_t)row * TOPK;
    const int*   ii = seli_g + (size_t)row * TOPK;
#pragma unroll 1
    for (int k = 0; k < TOPK; k++) {
        float val = vv[k];
        int idx = ii[k];
        float4 wv = ((const float4*)(WdT + (size_t)idx * IN_DIM))[t];
        acc.x += val * wv.x; acc.y += val * wv.y;
        acc.z += val * wv.z; acc.w += val * wv.w;
    }
    ((float4*)(Xrec + (size_t)row * IN_DIM))[t] = acc;
}

// ---------------------------------------------------------------- launch
extern "C" void kernel_launch(void* const* d_in, const int* in_sizes, int n_in,
                              void* d_out, int out_size, void* d_ws, size_t ws_size,
                              hipStream_t stream) {
    const float* x     = (const float*)d_in[0];
    const float* W_enc = (const float*)d_in[1];
    const float* b_enc = (const float*)d_in[2];
    const float* W_dec = (const float*)d_in[3];
    const float* b_dec = (const float*)d_in[4];

    char* ws = (char*)d_ws;
    unsigned short* x_bf  = (unsigned short*)(ws);                        // 16 MB
    unsigned short* we_bf = (unsigned short*)(ws + (size_t)16 * 1048576); // 32 MB
    float* wdT  = (float*)(ws + (size_t)48 * 1048576);                    // 64 MB
    float* selv = (float*)(ws + (size_t)112 * 1048576);                   // 1 MB
    int*   seli = (int*)  (ws + (size_t)113 * 1048576);                   // 1 MB

    float* xrec = (float*)d_out;                                          // [8192][1024]
    float* z    = (float*)d_out + (size_t)BATCH * IN_DIM;                 // [8192][16384]

    cvt_f32_bf16<<<(BATCH * IN_DIM / 4 + 255) / 256, 256, 0, stream>>>(x, x_bf, BATCH * IN_DIM / 4);
    cvt_f32_bf16<<<(LAT * IN_DIM / 4 + 255) / 256, 256, 0, stream>>>(W_enc, we_bf, LAT * IN_DIM / 4);
    transpose_wdec<<<dim3(LAT / 32, IN_DIM / 32), dim3(32, 8), 0, stream>>>(W_dec, wdT);
    encoder_wmma<<<dim3(LAT / 128, BATCH / 128), 256, 0, stream>>>(x_bf, we_bf, b_enc, z);
    topk_mask<<<BATCH, 256, 0, stream>>>(z, selv, seli);
    decoder_gather<<<BATCH, 256, 0, stream>>>(selv, seli, wdT, b_dec, xrec);
}
